// VectorQuantizer_48017734369463
// MI455X (gfx1250) — compile-verified, matrix operand order checked by reading, not run on hardware
//
#include <hip/hip_runtime.h>
#include <hip/hip_bf16.h>
#include <stdint.h>

// VQ nearest-codebook quantization for MI455X (gfx1250, wave32).
// d_in[0]: x (16,64,64,64) f32 ; d_in[1]: codebook (1024,64) f32
// d_out: [quantized 4,194,304 f32][indices 65,536 int32]
// d_ws : [bf16 codebook WMMA fragments 262,144 B][cb_norm 4,096 B]

typedef __attribute__((ext_vector_type(16))) __bf16 v16bf;
typedef __attribute__((ext_vector_type(8)))  float  v8f;

#define DIM     64
#define HW      4096   // H*W
#define NROWS   65536  // B*H*W
#define KCODES  1024
#define KT      64     // KCODES/16 code tiles
#define WS_NORM_OFF 262144  // bytes: 1024 codes * 64 dims * 2(hi,lo) * 2B

// K index mapping for 16-bit WMMA A/B fragments (ISA 7.12.2):
// lane half 0 holds K {0..7,16..23}+32f, half 1 holds K {8..15,24..31}+32f.
__device__ __forceinline__ int frag_k(int f, int half, int i) {
    return f * 32 + (i < 8 ? i + half * 8 : i + 8 + half * 8);
}

// ---------------------------------------------------------------------------
// Prep: pack codebook into bf16 hi/lo B-fragments + per-code squared norms.
// One wave per 16-code tile. Fragment q = kt*4 + f*2 + s (f: D-half, s: hi/lo),
// lane l's v16bf stored at element offset (q*32 + l)*16.
// ---------------------------------------------------------------------------
__global__ void vq_prep_kernel(const float* __restrict__ cb,
                               __bf16* __restrict__ wfrag,
                               float* __restrict__ cbn) {
    const int lane = threadIdx.x;      // 0..31
    const int kt   = blockIdx.x;       // 0..63
    const int col  = lane & 15;
    const int half = lane >> 4;
    const int code = kt * 16 + col;
    const float* src = cb + (size_t)code * DIM;

    float ss = 0.f;
    v16bf fh[2], fl[2];
#pragma unroll
    for (int f = 0; f < 2; ++f) {
#pragma unroll
        for (int i = 0; i < 16; ++i) {
            const int k = frag_k(f, half, i);
            const float v = src[k];
            ss += v * v;                       // each k covered once per (lane,l+16) pair
            const __bf16 h = (__bf16)v;
            fh[f][i] = h;
            fl[f][i] = (__bf16)(v - (float)h); // residual for error-compensated matmul
        }
    }
    v16bf* W = (v16bf*)wfrag;
#pragma unroll
    for (int f = 0; f < 2; ++f) {
        W[(size_t)(kt * 4 + f * 2 + 0) * 32 + lane] = fh[f];
        W[(size_t)(kt * 4 + f * 2 + 1) * 32 + lane] = fl[f];
    }
    const float tot = ss + __shfl_xor(ss, 16, 32);
    if (half == 0) cbn[code] = tot;
}

// ---------------------------------------------------------------------------
// Main: 256 threads = 8 waves, each wave owns 16 query rows (block: 128 rows).
// x tile staged to LDS (transposed, padded). Per code-tile: B fragments
// async-copied to double-buffered LDS, 6x v_wmma_f32_16x16x32_bf16
// (hi*hi + lo*hi + hi*lo per D-half), argmin tracked in registers.
// ---------------------------------------------------------------------------
__global__ __launch_bounds__(256)
void vq_main_kernel(const float* __restrict__ x,
                    const float* __restrict__ cb,
                    const __bf16* __restrict__ wfrag,
                    const float* __restrict__ cbn,
                    float* __restrict__ outq,
                    int* __restrict__ outidx) {
    __shared__ float xs[128 * 66];                 // 33,792 B, pad 66 vs banks
    __shared__ __align__(32) char bbuf[2][4096];   // double-buffered B fragments
    __shared__ int idxsh[128];

    const int tid  = threadIdx.x;
    const int lane = tid & 31;
    const int wave = tid >> 5;
    const int col  = lane & 15;
    const int half = lane >> 4;

    const int nBlock = blockIdx.x * 128;   // first global row of this block
    const int b   = nBlock >> 12;          // all 128 rows share one batch image
    const int hw0 = nBlock & 4095;

    // ---- stage x tile (rows nBlock..+127, all 64 dims) coalesced -> LDS ----
    for (int i = tid; i < 128 * 64; i += 256) {
        const int row = i & 127;           // consecutive lanes -> consecutive hw
        const int d   = i >> 7;
        xs[row * 66 + d] = x[(size_t)(b * 64 + d) * HW + hw0 + row];
    }

    // ---- kick off async copy of first B tile while we build A fragments ----
    {
        const unsigned ldsa = (unsigned)(uintptr_t)(&bbuf[0][tid * 16]);
        const char* g = (const char*)wfrag + (size_t)tid * 16;
        asm volatile("global_load_async_to_lds_b128 %0, %1, off"
                     :: "v"(ldsa), "v"(g) : "memory");
    }

    __syncthreads();  // xs visible to whole block

    // ---- per-wave A fragments (16 rows x 64 dims, hi/lo split) ----
    const int r = wave * 16 + col;         // local row this lane owns in A
    v16bf ah[2], al[2];
#pragma unroll
    for (int f = 0; f < 2; ++f) {
#pragma unroll
        for (int i = 0; i < 16; ++i) {
            const int k = frag_k(f, half, i);
            const float v = xs[r * 66 + k];
            const __bf16 h = (__bf16)v;
            ah[f][i] = h;
            al[f][i] = (__bf16)(v - (float)h);
        }
    }

    float best[8];
    int   bidx[8];
#pragma unroll
    for (int j = 0; j < 8; ++j) { best[j] = 3.4e38f; bidx[j] = 0; }

    int cur = 0;
    for (int kt = 0; kt < KT; ++kt) {
        if (kt + 1 < KT) {  // prefetch next tile into the other buffer
            const unsigned ldsa = (unsigned)(uintptr_t)(&bbuf[cur ^ 1][tid * 16]);
            const char* g = (const char*)wfrag + (size_t)(kt + 1) * 4096 + (size_t)tid * 16;
            asm volatile("global_load_async_to_lds_b128 %0, %1, off"
                         :: "v"(ldsa), "v"(g) : "memory");
            asm volatile("s_wait_asynccnt 0x1" ::: "memory");  // kt's copy done
        } else {
            asm volatile("s_wait_asynccnt 0x0" ::: "memory");
        }
        __syncthreads();  // bbuf[cur] ready for every wave

        const v16bf* B = (const v16bf*)(&bbuf[cur][0]);
        const v16bf b0h = B[0 * 32 + lane];
        const v16bf b0l = B[1 * 32 + lane];
        const v16bf b1h = B[2 * 32 + lane];
        const v16bf b1l = B[3 * 32 + lane];

        // two independent accumulation chains for ILP
        v8f acc0 = {}, acc1 = {};
        acc0 = __builtin_amdgcn_wmma_f32_16x16x32_bf16(false, ah[0], false, b0h, (short)0, acc0, false, false);
        acc1 = __builtin_amdgcn_wmma_f32_16x16x32_bf16(false, ah[1], false, b1h, (short)0, acc1, false, false);
        acc0 = __builtin_amdgcn_wmma_f32_16x16x32_bf16(false, al[0], false, b0h, (short)0, acc0, false, false);
        acc1 = __builtin_amdgcn_wmma_f32_16x16x32_bf16(false, al[1], false, b1h, (short)0, acc1, false, false);
        acc0 = __builtin_amdgcn_wmma_f32_16x16x32_bf16(false, ah[0], false, b0l, (short)0, acc0, false, false);
        acc1 = __builtin_amdgcn_wmma_f32_16x16x32_bf16(false, ah[1], false, b1l, (short)0, acc1, false, false);

        const float cn  = cbn[kt * 16 + col];
        const int  code = kt * 16 + col;
#pragma unroll
        for (int j = 0; j < 8; ++j) {
            const float dot = acc0[j] + acc1[j];
            const float s = fmaf(dot, -2.0f, cn);   // ||x||^2 constant per row -> drop
            const bool better = s < best[j];
            best[j] = better ? s : best[j];
            bidx[j] = better ? code : bidx[j];
        }
        __syncthreads();  // everyone done reading bbuf[cur] before refill
        cur ^= 1;
    }

    // ---- reduce argmin across the 16 lanes holding each row (stay in half) ----
#pragma unroll
    for (int m = 1; m < 16; m <<= 1) {
#pragma unroll
        for (int j = 0; j < 8; ++j) {
            const float ov = __shfl_xor(best[j], m, 32);
            const int   oi = __shfl_xor(bidx[j], m, 32);
            if (ov < best[j] || (ov == best[j] && oi < bidx[j])) {
                best[j] = ov; bidx[j] = oi;
            }
        }
    }

    if (col == 0) {  // lanes 0 and 16 hold rows j and j+8
#pragma unroll
        for (int j = 0; j < 8; ++j) {
            const int lrow = wave * 16 + j + half * 8;
            idxsh[lrow] = bidx[j];
            outidx[nBlock + lrow] = bidx[j];
        }
    }
    // intra-wave LDS producer/consumer; compiler inserts s_wait_dscnt

    // ---- gather winning codebook rows; coalesced along hw ----
    {
        const int lrow = wave * 16 + col;
        const int n_g  = nBlock + lrow;
        const int code = idxsh[lrow];
        const float* crow = cb + (size_t)code * DIM;
        const int hw = n_g & 4095;
#pragma unroll
        for (int t = 0; t < 32; ++t) {
            const int d = 2 * t + half;   // halves interleave over d
            outq[(size_t)(b * 64 + d) * HW + hw] = crow[d];
        }
    }
}

extern "C" void kernel_launch(void* const* d_in, const int* in_sizes, int n_in,
                              void* d_out, int out_size, void* d_ws, size_t ws_size,
                              hipStream_t stream) {
    (void)in_sizes; (void)n_in; (void)out_size; (void)ws_size;
    const float* x  = (const float*)d_in[0];
    const float* cb = (const float*)d_in[1];

    __bf16* wfrag = (__bf16*)d_ws;                            // 262,144 B
    float*  cbn   = (float*)((char*)d_ws + WS_NORM_OFF);      //   4,096 B

    float* outq   = (float*)d_out;                            // quantized
    int*   outidx = (int*)((float*)d_out + (size_t)NROWS * DIM / 1); // after 4,194,304 f32
    // NROWS*DIM == 16*64*64*64 quantized elements; indices follow as int32.

    vq_prep_kernel<<<dim3(KT), dim3(32), 0, stream>>>(cb, wfrag, cbn);
    vq_main_kernel<<<dim3(NROWS / 128), dim3(256), 0, stream>>>(x, cb, wfrag, cbn, outq, outidx);
}
